// BertBiLSTMCRF_52493090291985
// MI455X (gfx1250) — matrix operand order repaired and checked by reading, not compiled, not run on hardware
//
#include <hip/hip_runtime.h>

// BERT-BiLSTM-CRF for MI455X (gfx1250, wave32, WMMA bf16)
// Pipeline:
//   k_cvt    : f32 -> bf16 weight conversion (w_ih_f/b, w_hh_f/b)
//   k_xproj  : WMMA GEMM  G[dir][t][b][4H] = x(b,t) @ W_ih^T + bias   (M=16384,K=768,N=2048)
//              B tile staged with GLOBAL_LOAD_ASYNC_TO_LDS_B128 (ASYNCcnt),
//              next A slice prefetched with global_prefetch.
//   k_lstm   : 2 blocks (fwd/bwd), per-step WMMA recurrence h@W_hh^T + gates, LDS-resident h/c
//   k_tag    : emissions em = [hf;hb] @ w_tag^T + b_tag
//   k_llh/k_loss : CRF log-likelihood (logsumexp forward) -> scalar loss
//   k_dec    : CRF Viterbi decode -> preds

#define B_  32
#define T_  512
#define D_  768
#define H_  256
#define G4_ 1024     // 4*H
#define L_  9

typedef __attribute__((ext_vector_type(16))) __bf16        v16bf;
typedef __attribute__((ext_vector_type(8)))  float         v8f;
typedef __attribute__((ext_vector_type(4)))  unsigned int  u32x4;
typedef __attribute__((ext_vector_type(4)))  float         f32x4;

union Frag {
    v16bf        v;
    unsigned int u[8];
    u32x4        q[2];
};

__device__ __forceinline__ unsigned short f2bf(float f) {
    unsigned u = __float_as_uint(f);
    u += 0x7FFFu + ((u >> 16) & 1u);        // round-to-nearest-even
    return (unsigned short)(u >> 16);
}
__device__ __forceinline__ float bf2f(unsigned short h) {
    return __uint_as_float(((unsigned)h) << 16);
}
__device__ __forceinline__ float sigm(float x) { return 1.0f / (1.0f + expf(-x)); }

__device__ __forceinline__ v8f wmma_bf16(const Frag& a, const Frag& b, v8f c) {
    // D = A(16x32 bf16) x B(32x16 bf16) + C(16x16 f32)
    return __builtin_amdgcn_wmma_f32_16x16x32_bf16(false, a.v, false, b.v,
                                                   (short)0, c, false, false);
}

// Build A fragment (16x32 bf16, contiguous-k layout, same k order as B) from a row-major bf16 row pointer.
// Lane (half,lr): element pair 2v,2v+1 holds
//   K = 2v + 16*half (+k0)
// -> one contiguous 32B run per lane; compiler merges into ds_load_b128.
__device__ __forceinline__ void build_a(Frag& a, const unsigned short* rowbase,
                                        int k0, int hlf) {
#pragma unroll
    for (int v = 0; v < 8; ++v) {
        int kk = k0 + 16 * hlf + 2 * v;
        a.u[v] = *(const unsigned int*)(rowbase + kk);
    }
}

// Async global -> LDS copy of 16 bytes per lane (CDNA5 GLOBAL_LOAD_ASYNC_TO_LDS_B128).
// ldsoff = low 32 bits of the generic shared-memory address;
// gaddr = 64-bit global address (GV mode, saddr = off).
__device__ __forceinline__ void async_copy_b128(void* lds_dst, const void* gsrc) {
    unsigned ldsoff = (unsigned)(size_t)lds_dst;
    unsigned long long gaddr = (unsigned long long)(size_t)gsrc;
    asm volatile("global_load_async_to_lds_b128 %0, %1, off"
                 :
                 : "v"(ldsoff), "v"(gaddr)
                 : "memory");
}
__device__ __forceinline__ void wait_asynccnt0() {
    asm volatile("s_wait_asynccnt 0x0" ::: "memory");
}

// ---------------------------------------------------------------- weight cvt
__global__ void k_cvt(const float* __restrict__ src, unsigned short* __restrict__ dst, int n) {
    int i = blockIdx.x * 256 + threadIdx.x;
    if (i < n) dst[i] = f2bf(src[i]);
}

// --------------------------------------------------------- input projections
// G[(dir*T + t)*B*4H + b*4H + g] = x(b,t) . W_ih[dir][g,:] + bias[dir][g]
// Block: 512 thr = 16 waves in 4x4; macro tile 128(M) x 128(N); K-step 32.
__global__ __launch_bounds__(512) void k_xproj(
    const float* __restrict__ X,                // (B*T, D) rows m = b*T + t
    const unsigned short* __restrict__ Wf,      // (4H, D) bf16
    const unsigned short* __restrict__ Wb,      // (4H, D) bf16
    const float* __restrict__ biasf,
    const float* __restrict__ biasb,
    float* __restrict__ G)
{
    __shared__ __align__(16) unsigned short A_s[128][32];   // [m][k]
    __shared__ __align__(16) unsigned short B_s[128][32];   // [n][k]

    const int tid = threadIdx.x;
    const int n0  = blockIdx.x * 128;           // over N = 2048
    const int m0  = blockIdx.y * 128;           // over M = 16384
    const unsigned short* W = (n0 < G4_) ? Wf : Wb;
    const int nbase = n0 & (G4_ - 1);
    const int wv = tid >> 5, lane = tid & 31, hlf = lane >> 4, lr = lane & 15;
    const int wm = wv >> 2, wn = wv & 3;        // wave -> 32x32 sub-tile

    v8f acc[2][2];
#pragma unroll
    for (int mt = 0; mt < 2; ++mt)
#pragma unroll
        for (int nt = 0; nt < 2; ++nt)
#pragma unroll
            for (int r = 0; r < 8; ++r) acc[mt][nt][r] = 0.0f;

    const int browa = tid >> 2, bsega = tid & 3;          // B-tile async copy slot

    for (int k0 = 0; k0 < D_; k0 += 32) {
        __syncthreads();
        // B tile: 128 rows x 32 k bf16, async global->LDS (1 x b128 per thread),
        // overlapped with the A-tile f32->bf16 conversion below.
        async_copy_b128(&B_s[browa][bsega * 8],
                        W + (size_t)(nbase + browa) * D_ + k0 + bsega * 8);

        // A tile: 128 rows x 32 k, f32 -> bf16 (1024 float4 loads, 2/thread)
#pragma unroll
        for (int i = 0; i < 2; ++i) {
            int idx = tid + i * 512;
            int row = idx >> 3, seg = idx & 7;
            const float* xp = X + (size_t)(m0 + row) * D_ + k0 + seg * 4;
            f32x4 v = *(const f32x4*)xp;
            if (k0 + 32 < D_) __builtin_prefetch(xp + 32, 0, 0);   // next K slice
            A_s[row][seg * 4 + 0] = f2bf(v[0]);
            A_s[row][seg * 4 + 1] = f2bf(v[1]);
            A_s[row][seg * 4 + 2] = f2bf(v[2]);
            A_s[row][seg * 4 + 3] = f2bf(v[3]);
        }
        wait_asynccnt0();            // barrier does not cover ASYNCcnt
        __syncthreads();

        Frag a[2];
        build_a(a[0], &A_s[wm * 32 + 0 * 16 + lr][0], 0, hlf);
        build_a(a[1], &A_s[wm * 32 + 1 * 16 + lr][0], 0, hlf);
#pragma unroll
        for (int nt = 0; nt < 2; ++nt) {
            Frag bfrag;   // B layout: lane lr = column, 16 contiguous K at 16*hlf
            const unsigned short* bp = &B_s[wn * 32 + nt * 16 + lr][16 * hlf];
            bfrag.q[0] = *(const u32x4*)(bp);
            bfrag.q[1] = *(const u32x4*)(bp + 8);
            acc[0][nt] = wmma_bf16(a[0], bfrag, acc[0][nt]);
            acc[1][nt] = wmma_bf16(a[1], bfrag, acc[1][nt]);
        }
    }

    // Epilogue: C layout M = r + 8*hlf, N = lr
#pragma unroll
    for (int mt = 0; mt < 2; ++mt)
#pragma unroll
        for (int nt = 0; nt < 2; ++nt)
#pragma unroll
            for (int r = 0; r < 8; ++r) {
                int m  = m0 + wm * 32 + mt * 16 + r + 8 * hlf;
                int gc = n0 + wn * 32 + nt * 16 + lr;
                int dir = gc >> 10, g = gc & (G4_ - 1);
                int b = m >> 9, t = m & (T_ - 1);
                float bias = dir ? biasb[g] : biasf[g];
                G[(((size_t)dir * T_ + t) * B_ + b) * G4_ + g] = acc[mt][nt][r] + bias;
            }
}

// ---------------------------------------------------------------- recurrence
// 2 blocks (dir), 512 thr = 16 waves. Per step: R = h @ W_hh^T via WMMA
// (M=32, N=1024, K=256; wave w owns columns [64w,64w+64)), gates in LDS,
// pointwise LSTM cell, h kept in LDS as bf16 for the next step's A-frags.
__global__ __launch_bounds__(512) void k_lstm(
    const unsigned short* __restrict__ Whf,    // (4H, H) bf16
    const unsigned short* __restrict__ Whb,
    const float* __restrict__ G,               // (2, T, B, 4H)
    unsigned short* __restrict__ H)            // (2, T, B, H) bf16
{
    extern __shared__ char smem[];
    float*          gates = (float*)smem;                               // 32x1024 f32
    float*          cst   = (float*)(smem + B_ * G4_ * 4);              // 32x256  f32
    unsigned short* hbf   = (unsigned short*)(smem + B_ * G4_ * 4 + B_ * H_ * 4); // 32x256 bf16

    const int dir = blockIdx.x;
    const unsigned short* W = dir ? Whb : Whf;
    const float* Gd = G + (size_t)dir * T_ * B_ * G4_;
    unsigned short* Hd = H + (size_t)dir * T_ * B_ * H_;

    const int tid = threadIdx.x, wv = tid >> 5, lane = tid & 31;
    const int hlf = lane >> 4, lr = lane & 15;

    for (int i = tid; i < B_ * H_; i += 512) { cst[i] = 0.0f; hbf[i] = 0; }
    __syncthreads();

    for (int s = 0; s < T_; ++s) {
        const int t = dir ? (T_ - 1 - s) : s;

        v8f acc[2][4];
#pragma unroll
        for (int mt = 0; mt < 2; ++mt)
#pragma unroll
            for (int nt = 0; nt < 4; ++nt)
#pragma unroll
                for (int r = 0; r < 8; ++r) acc[mt][nt][r] = 0.0f;

        for (int ks = 0; ks < 8; ++ks) {        // K = 256 in steps of 32
            const int k0 = ks * 32;
            Frag a0, a1;
            build_a(a0, hbf + (size_t)(0 * 16 + lr) * H_, k0, hlf);
            build_a(a1, hbf + (size_t)(16 + lr) * H_, k0, hlf);
#pragma unroll
            for (int nt = 0; nt < 4; ++nt) {
                Frag bfrag;  // W row n, 16 contiguous k at k0+16*hlf (L2-resident)
                const unsigned short* wp =
                    W + (size_t)(wv * 64 + nt * 16 + lr) * H_ + k0 + 16 * hlf;
                bfrag.q[0] = *(const u32x4*)(wp);
                bfrag.q[1] = *(const u32x4*)(wp + 8);
                acc[0][nt] = wmma_bf16(a0, bfrag, acc[0][nt]);
                acc[1][nt] = wmma_bf16(a1, bfrag, acc[1][nt]);
            }
        }

        const float* Gt = Gd + (size_t)t * B_ * G4_;
#pragma unroll
        for (int mt = 0; mt < 2; ++mt)
#pragma unroll
            for (int nt = 0; nt < 4; ++nt)
#pragma unroll
                for (int r = 0; r < 8; ++r) {
                    int b = mt * 16 + r + 8 * hlf;
                    int n = wv * 64 + nt * 16 + lr;
                    gates[b * G4_ + n] = acc[mt][nt][r] + Gt[b * G4_ + n];
                }
        __syncthreads();

        for (int i = tid; i < B_ * H_; i += 512) {
            int b = i >> 8, j = i & (H_ - 1);
            float ig = gates[b * G4_ + j];
            float fg = gates[b * G4_ + H_ + j];
            float gg = gates[b * G4_ + 2 * H_ + j];
            float og = gates[b * G4_ + 3 * H_ + j];
            float cn = sigm(fg) * cst[i] + sigm(ig) * tanhf(gg);
            float hv = sigm(og) * tanhf(cn);
            cst[i] = cn;
            unsigned short hb16 = f2bf(hv);
            hbf[i] = hb16;
            Hd[(size_t)t * B_ * H_ + i] = hb16;
        }
        __syncthreads();
    }
}

// ----------------------------------------------------------------- emissions
// em[b,t,l] = [hf(t,b,:); hb(t,b,:)] . w_tag[l,:] + b_tag[l]
__global__ __launch_bounds__(256) void k_tag(
    const unsigned short* __restrict__ H,
    const float* __restrict__ wtag,            // (L, 2H)
    const float* __restrict__ btag,
    float* __restrict__ em)                    // (B, T, L)
{
    __shared__ float wt[L_ * 2 * H_];
    __shared__ float bt[L_];
    const int tid = threadIdx.x;
    for (int i = tid; i < L_ * 2 * H_; i += 256) wt[i] = wtag[i];
    if (tid < L_) bt[tid] = btag[tid];
    __syncthreads();

    const int warp = tid >> 5, lane = tid & 31;
    const int m = blockIdx.x * 8 + warp;       // m = b*T + t
    const int b = m >> 9, t = m & (T_ - 1);
    const unsigned short* hf = H + ((size_t)t * B_ + b) * H_;
    const unsigned short* hb = H + ((size_t)(T_ + t) * B_ + b) * H_;
    if (lane < L_) {
        float acc = bt[lane];
        for (int j = 0; j < H_; ++j) acc += bf2f(hf[j]) * wt[lane * 2 * H_ + j];
        for (int j = 0; j < H_; ++j) acc += bf2f(hb[j]) * wt[lane * 2 * H_ + H_ + j];
        em[(size_t)m * L_ + lane] = acc;
    }
}

// ------------------------------------------------------------- CRF likelihood
__global__ __launch_bounds__(32) void k_llh(
    const float* __restrict__ em, const int* __restrict__ mask,
    const int* __restrict__ labels, const float* __restrict__ startv,
    const float* __restrict__ endv, const float* __restrict__ trans,
    float* __restrict__ llh)
{
    __shared__ float tr[L_ * L_];
    const int b = blockIdx.x, j = threadIdx.x;
    for (int i = j; i < L_ * L_; i += 32) tr[i] = trans[i];
    __syncthreads();

    const float* emb = em + (size_t)b * T_ * L_;
    const int* mk = mask + b * T_;
    const int jj = (j < L_) ? j : 0;
    float sc = (j < L_) ? (startv[j] + emb[j]) : -1e30f;

    for (int t = 1; t < T_; ++t) {
        float vals[L_];
        float mx = -1e30f;
#pragma unroll
        for (int i = 0; i < L_; ++i) {
            float si = __shfl(sc, i, 32) + tr[i * L_ + jj];
            vals[i] = si;
            mx = fmaxf(mx, si);
        }
        float ssum = 0.0f;
#pragma unroll
        for (int i = 0; i < L_; ++i) ssum += expf(vals[i] - mx);
        float nxt = mx + logf(ssum) + emb[t * L_ + jj];
        if (mk[t] > 0 && j < L_) sc = nxt;
    }
    float v = (j < L_) ? (sc + endv[j]) : -1e30f;
    float mx = -1e30f;
#pragma unroll
    for (int i = 0; i < L_; ++i) mx = fmaxf(mx, __shfl(v, i, 32));
    float ssum = 0.0f;
#pragma unroll
    for (int i = 0; i < L_; ++i) ssum += expf(__shfl(v, i, 32) - mx);
    float denom = mx + logf(ssum);

    if (j == 0) {   // numerator (serial, tiny)
        int t0 = labels[b * T_];
        int prev = (t0 == -100) ? 0 : t0;
        float num = startv[prev] + emb[prev];
        int msum = (mk[0] > 0) ? 1 : 0;
        for (int t = 1; t < T_; ++t) {
            int tg = labels[b * T_ + t]; if (tg == -100) tg = 0;
            num += (tr[prev * L_ + tg] + emb[t * L_ + tg]) * (float)mk[t];
            prev = tg;
            msum += (mk[t] > 0) ? 1 : 0;
        }
        int lt = labels[b * T_ + (msum - 1)]; if (lt == -100) lt = 0;
        num += endv[lt];
        llh[b] = num - denom;
    }
}

__global__ void k_loss(const float* __restrict__ llh, float* __restrict__ out) {
    float v = llh[threadIdx.x];
#pragma unroll
    for (int off = 16; off > 0; off >>= 1) v += __shfl_down(v, off, 32);
    if (threadIdx.x == 0) out[0] = -v / (float)B_;
}

// --------------------------------------------------------------- CRF decode
__global__ __launch_bounds__(32) void k_dec(
    const float* __restrict__ em, const int* __restrict__ mask,
    const float* __restrict__ startv, const float* __restrict__ endv,
    const float* __restrict__ trans, float* __restrict__ preds)
{
    __shared__ float tr[L_ * L_];
    __shared__ unsigned char bp[T_ - 1][L_];
    const int b = blockIdx.x, j = threadIdx.x;
    for (int i = j; i < L_ * L_; i += 32) tr[i] = trans[i];
    __syncthreads();

    const float* emb = em + (size_t)b * T_ * L_;
    const int* mk = mask + b * T_;
    const int jj = (j < L_) ? j : 0;
    float sc = (j < L_) ? (startv[j] + emb[j]) : -1e30f;

    for (int t = 1; t < T_; ++t) {
        float best = -1e30f; int arg = 0;
#pragma unroll
        for (int i = 0; i < L_; ++i) {
            float si = __shfl(sc, i, 32) + tr[i * L_ + jj];
            if (si > best) { best = si; arg = i; }
        }
        float nxt = best + emb[t * L_ + jj];
        bool keep = mk[t] > 0;
        if (j < L_) {
            bp[t - 1][j] = (unsigned char)(keep ? arg : j);
            if (keep) sc = nxt;
        }
    }
    __syncthreads();
    float v = (j < L_) ? (sc + endv[j]) : -1e30f;
    float best = -1e30f; int arg = 0;
#pragma unroll
    for (int i = 0; i < L_; ++i) {
        float si = __shfl(v, i, 32);
        if (si > best) { best = si; arg = i; }
    }
    if (j == 0) {
        int cur = arg;
        preds[(size_t)b * T_ + (T_ - 1)] = (float)cur;
        for (int s = T_ - 2; s >= 0; --s) {
            cur = bp[s][cur];
            preds[(size_t)b * T_ + s] = (float)cur;
        }
    }
}

// ------------------------------------------------------------------- launch
extern "C" void kernel_launch(void* const* d_in, const int* in_sizes, int n_in,
                              void* d_out, int out_size, void* d_ws, size_t ws_size,
                              hipStream_t stream)
{
    (void)in_sizes; (void)n_in; (void)out_size; (void)ws_size;

    const float* seq    = (const float*)d_in[0];
    const int*   amask  = (const int*)d_in[1];
    const int*   labels = (const int*)d_in[2];
    const float* wihf   = (const float*)d_in[3];
    const float* whhf   = (const float*)d_in[4];
    const float* bfv    = (const float*)d_in[5];
    const float* wihb   = (const float*)d_in[6];
    const float* whhb   = (const float*)d_in[7];
    const float* bbv    = (const float*)d_in[8];
    const float* wtag   = (const float*)d_in[9];
    const float* btag   = (const float*)d_in[10];
    const float* startv = (const float*)d_in[11];
    const float* endv   = (const float*)d_in[12];
    const float* trans  = (const float*)d_in[13];

    // workspace layout (bytes)
    char* ws = (char*)d_ws;
    unsigned short* Wihf = (unsigned short*)(ws + 0);          // 1024*768*2
    unsigned short* Wihb = (unsigned short*)(ws + 1572864);    // 1024*768*2
    unsigned short* Whhf = (unsigned short*)(ws + 3145728);    // 1024*256*2
    unsigned short* Whhb = (unsigned short*)(ws + 3670016);    // 1024*256*2
    float*          G    = (float*)(ws + 4194304);             // 2*512*32*1024*4 = 128MB
    unsigned short* H    = (unsigned short*)(ws + 138412032);  // 2*512*32*256*2  = 16MB
    float*          llh  = (float*)(ws + 155189248);           // 32*4

    float* em    = (float*)d_out;        // (B,T,L) = 147456
    float* loss  = em + 147456;          // scalar
    float* preds = em + 147457;          // (B,T) as floats

    k_cvt<<<dim3((786432 + 255) / 256), dim3(256), 0, stream>>>(wihf, Wihf, 786432);
    k_cvt<<<dim3((786432 + 255) / 256), dim3(256), 0, stream>>>(wihb, Wihb, 786432);
    k_cvt<<<dim3((262144 + 255) / 256), dim3(256), 0, stream>>>(whhf, Whhf, 262144);
    k_cvt<<<dim3((262144 + 255) / 256), dim3(256), 0, stream>>>(whhb, Whhb, 262144);

    k_xproj<<<dim3(16, 128), dim3(512), 0, stream>>>(seq, Wihf, Wihb, bfv, bbv, G);

    size_t lds_bytes = (size_t)B_ * G4_ * 4 + (size_t)B_ * H_ * 4 + (size_t)B_ * H_ * 2; // 180224
    k_lstm<<<dim3(2), dim3(512), lds_bytes, stream>>>(Whhf, Whhb, G, H);

    k_tag<<<dim3(2048), dim3(256), 0, stream>>>(H, wtag, btag, em);

    k_llh<<<dim3(32), dim3(32), 0, stream>>>(em, amask, labels, startv, endv, trans, llh);
    k_loss<<<dim3(1), dim3(32), 0, stream>>>(llh, loss);
    k_dec<<<dim3(32), dim3(32), 0, stream>>>(em, amask, startv, endv, trans, preds);
}
